// ScaleDotProductAttention_69483980915008
// MI455X (gfx1250) — matrix operand order repaired and checked by reading, not compiled, not run on hardware
//
#include <hip/hip_runtime.h>

// CDNA5 / gfx1250 fused attention (reshape-K variant + softmax over heads).
// Both GEMMs use V_WMMA_F32_16X16X32_BF16 (bf16 operands, f32 accumulate):
// the problem is HBM-bound (~1.2 GB mandatory traffic -> ~52us at 23.3 TB/s),
// and bf16 WMMA keeps matrix compute far under that floor, unlike f32 K=4 WMMA.
//
// Round 4: async global->LDS staging of the Q tile via
// GLOBAL_LOAD_ASYNC_TO_LDS_B128. Clang's round-3 diagnostic showed the
// builtin's signature: (int4 AS1*, int4 AS3*, imm offset, imm cpol) — pass
// exactly-typed address-space pointers.

typedef __attribute__((ext_vector_type(16))) __bf16 v16bf;
typedef __attribute__((ext_vector_type(8)))  float  v8f;
typedef __attribute__((ext_vector_type(4)))  float  v4f;
typedef __attribute__((ext_vector_type(4)))  int    v4i;

typedef __attribute__((address_space(1))) v4i* gas_v4i;   // global (AS1)
typedef __attribute__((address_space(3))) v4i* las_v4i;   // LDS (AS3)

#define B_  2
#define H_  16
#define S_  2048
#define D_  128
#define SCALE 0.08838834764831845f   // 1/sqrt(128)

static __device__ __forceinline__ __bf16 f2bf(float x) { return (__bf16)x; }

#if defined(__has_builtin)
#if __has_builtin(__builtin_amdgcn_global_load_async_to_lds_b128)
#define USE_ASYNC_LDS 1
#endif
#endif

// ---------------------------------------------------------------------------
// Kernel A: score = softmax_over_heads( mask==0 ? -1e-12 : (Q @ K_reshaped)/sqrt(d) )
// Block: 256 threads (8 waves). Wave w owns the 16x16 tile at
// (i0, blockIdx.y*128 + w*16) and iterates all 16 heads, holding one v8f
// C-fragment per head in registers. Because the WMMA C layout is identical
// for every head, the softmax over the head axis is lane-local (no LDS).
// ---------------------------------------------------------------------------
__global__ __launch_bounds__(256, 1)
void attn_score_softmax_kernel(const float* __restrict__ q,
                               const float* __restrict__ k,
                               const int*   __restrict__ mask,
                               float* __restrict__ score_out)
{
    __shared__ float qs[16 * D_];          // 8 KB: Q tile for current head

    const int tid  = threadIdx.x;
    const int wave = tid >> 5;
    const int lane = tid & 31;
    const int half = lane >> 4;            // 0: lanes 0-15, 1: lanes 16-31
    const int n    = lane & 15;

    const int i0 = blockIdx.x * 16;
    const int j0 = blockIdx.y * 128 + wave * 16;
    const int b  = blockIdx.z;

    const float* qb = q + (size_t)b * H_ * S_ * D_;
    const float* kb = k + (size_t)b * H_ * S_ * D_;

    v8f acc[H_];

    #pragma unroll
    for (int h = 0; h < H_; ++h) {
        // ---- stage Q tile (16 x 128 f32) for head h into LDS ----
        {
            const float* qh = qb + (size_t)h * S_ * D_ + (size_t)i0 * D_;
            const int off = tid * 8;                   // 2048 floats / 256 thr
#if defined(USE_ASYNC_LDS)
            // Direct global->LDS DMA, 2 x 16B per thread, tracked by ASYNCcnt.
            __builtin_amdgcn_global_load_async_to_lds_b128(
                (gas_v4i)(qh + off), (las_v4i)(qs + off), 0, 0);
            __builtin_amdgcn_global_load_async_to_lds_b128(
                (gas_v4i)(qh + off + 4), (las_v4i)(qs + off + 4), 0, 0);
            asm volatile("s_wait_asynccnt 0x0" ::: "memory");
#else
            v4f x0 = *(const v4f*)(qh + off);
            v4f x1 = *(const v4f*)(qh + off + 4);
            *(v4f*)(qs + off)     = x0;
            *(v4f*)(qs + off + 4) = x1;
#endif
        }
        __syncthreads();

        const float* kh  = kb + (size_t)h * S_ * D_;   // reshape view [128,2048]
        const float* qrow = qs + n * D_;               // A row M = lane&15

        v8f c = {0.f, 0.f, 0.f, 0.f, 0.f, 0.f, 0.f, 0.f};
        #pragma unroll
        for (int kc = 0; kc < 4; ++kc) {
            // A 16x32 fragment from LDS: per-lane K runs [half*8, +8) -> elems
            // 0..7 and [16+half*8, +8) -> elems 8..15 within this 32-K chunk.
            v16bf afrag;
            {
                const float* row = qrow + kc * 32;
                #pragma unroll
                for (int t = 0; t < 8; ++t) {
                    afrag[t]     = f2bf(row[half * 8 + t]);
                    afrag[8 + t] = f2bf(row[16 + half * 8 + t]);
                }
            }
            // B 32x16 fragment from the reshape-view K (row-major [128,2048]):
            // column N = lane&15, K runs [half*8, +8) and [16+half*8, +8) (matches A).
            v16bf bfrag;
            {
                const int kbase = kc * 32 + half * 8;
                #pragma unroll
                for (int t = 0; t < 16; ++t) {
                    bfrag[t] = f2bf(kh[(size_t)(kbase + t + (t >= 8 ? 8 : 0)) * S_ + (j0 + n)]);
                }
            }
            c = __builtin_amdgcn_wmma_f32_16x16x32_bf16(
                    false, afrag, false, bfrag, (short)0, c, false, false);
        }
        acc[h] = c;
        __syncthreads();   // qs is rewritten next head iteration
    }

    // ---- mask + softmax over the 16 heads (lane-local), write score ----
    const int jcol = j0 + n;
    #pragma unroll
    for (int e = 0; e < 8; ++e) {
        const int irow = i0 + e + half * 8;            // C layout: M = e + 8*half
        const int mv   = mask[(size_t)irow * S_ + jcol];

        float x[H_];
        float m = -3.402823e38f;
        #pragma unroll
        for (int h = 0; h < H_; ++h) {
            float v = (mv == 0) ? -1e-12f : acc[h][e] * SCALE;
            x[h] = v;
            m = fmaxf(m, v);
        }
        float sum = 0.f;
        #pragma unroll
        for (int h = 0; h < H_; ++h) {
            float ex = __expf(x[h] - m);
            x[h] = ex;
            sum += ex;
        }
        const float inv = 1.0f / sum;
        #pragma unroll
        for (int h = 0; h < H_; ++h) {
            score_out[(((size_t)b * H_ + h) * S_ + irow) * S_ + jcol] = x[h] * inv;
        }
    }
}

// ---------------------------------------------------------------------------
// Kernel B: out[b,h] = score[b,h] (2048x2048) @ V[b,h] (2048x128)
// Block: 256 threads (8 waves); wave w owns out columns [w*16, w*16+16).
// ---------------------------------------------------------------------------
__global__ __launch_bounds__(256, 1)
void attn_out_kernel(const float* __restrict__ score,
                     const float* __restrict__ v,
                     float* __restrict__ out)
{
    const int tid  = threadIdx.x;
    const int wave = tid >> 5;
    const int lane = tid & 31;
    const int half = lane >> 4;
    const int n    = lane & 15;

    const int i0 = blockIdx.x * 16;
    const int bh = blockIdx.y;
    const int j0 = wave * 16;

    const float* sb = score + (size_t)bh * S_ * S_;
    const float* vb = v     + (size_t)bh * S_ * D_;
    const float* arow = sb + (size_t)(i0 + n) * S_;    // A row M = lane&15

    v8f c = {0.f, 0.f, 0.f, 0.f, 0.f, 0.f, 0.f, 0.f};

    for (int kk = 0; kk < S_; kk += 32) {
        // A fragment: two contiguous 8-float runs (f32 score -> bf16)
        v16bf afrag;
        {
            const float* p = arow + kk;
            v4f r0 = *(const v4f*)(p + half * 8);
            v4f r1 = *(const v4f*)(p + half * 8 + 4);
            v4f r2 = *(const v4f*)(p + 16 + half * 8);
            v4f r3 = *(const v4f*)(p + 16 + half * 8 + 4);
            #pragma unroll
            for (int t = 0; t < 4; ++t) {
                afrag[t]      = f2bf(r0[t]);
                afrag[4 + t]  = f2bf(r1[t]);
                afrag[8 + t]  = f2bf(r2[t]);
                afrag[12 + t] = f2bf(r3[t]);
            }
        }

        // B fragment: V rows kbase..kbase+7 and kbase+16..kbase+23, column j0+n (stride D_)
        v16bf bfrag;
        const int kbase = kk + half * 8;
        #pragma unroll
        for (int t = 0; t < 16; ++t) {
            bfrag[t] = f2bf(vb[(size_t)(kbase + t + (t >= 8 ? 8 : 0)) * D_ + (j0 + n)]);
        }

        if (kk + 32 < S_) {
            __builtin_prefetch(vb + (size_t)(kk + 32) * D_ + (j0 + n), 0, 1);
        }

        c = __builtin_amdgcn_wmma_f32_16x16x32_bf16(
                false, afrag, false, bfrag, (short)0, c, false, false);
    }

    #pragma unroll
    for (int e = 0; e < 8; ++e) {
        const int irow = i0 + e + half * 8;
        out[((size_t)bh * S_ + irow) * D_ + (j0 + n)] = c[e];
    }
}

// ---------------------------------------------------------------------------
extern "C" void kernel_launch(void* const* d_in, const int* in_sizes, int n_in,
                              void* d_out, int out_size, void* d_ws, size_t ws_size,
                              hipStream_t stream) {
    (void)in_sizes; (void)n_in; (void)out_size; (void)d_ws; (void)ws_size;

    const float* q    = (const float*)d_in[0];
    const float* k    = (const float*)d_in[1];
    const float* v    = (const float*)d_in[2];
    const int*   mask = (const int*)  d_in[3];

    float* out   = (float*)d_out;                       // [B,H,S,D] first
    float* score = out + (size_t)B_ * H_ * S_ * D_;     // then [B,H,S,S]

    dim3 gridA(S_ / 16, S_ / 128, B_);                  // 128 x 16 x 2 blocks
    attn_score_softmax_kernel<<<gridA, 256, 0, stream>>>(q, k, mask, score);

    dim3 gridB(S_ / 16, B_ * H_);                       // 128 x 32 blocks
    attn_out_kernel<<<gridB, 256, 0, stream>>>(score, v, out);
}
